// QuantizedSimpleSSM_15530601742692
// MI455X (gfx1250) — matrix operand order repaired and checked by reading, not compiled, not run on hardware
//
#include <hip/hip_runtime.h>
#include <hip/hip_bf16.h>

typedef __attribute__((ext_vector_type(16))) _Float16 v16h;
typedef __attribute__((ext_vector_type(8)))  _Float16 v8h;
typedef __attribute__((ext_vector_type(8)))  float    v8f;

#define BZ   32
#define DIN  512
#define LEN  512
#define NN   512
#define DOUT 512
#define RR   (LEN * BZ)   // 16384 rows (l*BZ + b)

// ---- reference fake-fp8 quantizer, branch-free integer-exponent version ----
// t = |x| + 1e-8 is always a normal f32, so floor(log2(t)) == IEEE exponent(t),
// and 2^e / 2^(3-e) are exact bit patterns. Matches the log2f/exp2f reference
// (rintf == round-half-even == jnp.round).
__device__ __forceinline__ float qdev(float x) {
    float ax = fabsf(x);
    float t  = ax + 1e-8f;
    int e = ((__float_as_int(t) >> 23) & 0xff) - 127;   // floor(log2(t))
    e = e < -7 ? -7 : (e > 7 ? 7 : e);
    float p    = __int_as_float((e + 127) << 23);       // 2^e
    float pinv = __int_as_float((3 - e + 127) << 23);   // 2^(3-e) = 8/p
    float m = rintf(fmaf(ax, pinv, -8.0f)) * 0.125f;    // round((ax/p - 1)*8)/8
    float s = (x > 0.0f) ? 1.0f : ((x < 0.0f) ? -1.0f : 0.0f);
    return s * (1.0f + m) * p;
}

// ---- elementwise quantize weight matrix to f16, keep original row-major layout ----
__global__ void quant_kernel(const float* __restrict__ src, _Float16* __restrict__ dst, int n) {
    int i = blockIdx.x * blockDim.x + threadIdx.x;
    if (i < n) dst[i] = (_Float16)qdev(src[i]);
}

__global__ void quantA_kernel(const float* __restrict__ src, float* __restrict__ dst, int n) {
    int i = blockIdx.x * blockDim.x + threadIdx.x;
    if (i < n) dst[i] = qdev(src[i]);
}

// ---- u (BZ, DIN, LEN) f32  ->  Urows [(l*BZ+b)][d] f32, LDS 32x32 tiles ----
__global__ void transpose_u_kernel(const float* __restrict__ u, float* __restrict__ Urows) {
    __shared__ float tile[32][33];
    int l0 = blockIdx.x * 32, d0 = blockIdx.y * 32, b = blockIdx.z;
    int tx = threadIdx.x, ty = threadIdx.y;            // (32, 8)
    const float* ub = u + (size_t)b * DIN * LEN;
#pragma unroll
    for (int i = 0; i < 4; ++i) {
        int d = d0 + ty + i * 8;
        tile[ty + i * 8][tx] = ub[(size_t)d * LEN + l0 + tx];
    }
    __syncthreads();
#pragma unroll
    for (int i = 0; i < 4; ++i) {
        int l = l0 + ty + i * 8;
        Urows[((size_t)l * BZ + b) * DIN + d0 + tx] = tile[tx][ty + i * 8];
    }
}

// ---- WMMA GEMM: Cout(RR x Ncols) = A(RR x K, f32) * Wq^T, Wq (Ncols x K, f16 row-major)
// 256 threads = 8 waves; block tile 64(M) x 64(N); each wave: two 16x16 M-tiles; K-step 32.
// Double-buffered LDS; weight tile staged [n][k]-major via GLOBAL_LOAD_ASYNC_TO_LDS_B128
// (ASYNCcnt) so the DMA for tile k+1 overlaps the WMMAs of tile k.
template<bool HAS_ADD, bool DO_QUANT>
__global__ __launch_bounds__(256)
void wmma_gemm_kernel(const float* __restrict__ A, const _Float16* __restrict__ Wq,
                      float* __restrict__ Cout, const float* __restrict__ addend,
                      int K, int Ncols) {
    __shared__ _Float16 ldsA[2][64][40];   // [buf][m-row][k], 80B rows (b128-aligned)
    __shared__ _Float16 ldsW[2][64][40];   // [buf][n-row][k]

    const int tid   = threadIdx.x;
    const int lane  = tid & 31;
    const int wid   = tid >> 5;         // 0..7
    const int tm    = wid >> 2;         // 0..1
    const int tn    = wid & 3;          // 0..3
    const int half  = lane >> 4;        // 0/1
    const int l16   = lane & 15;

    const int rbase = blockIdx.x * 64;
    const int nbase = blockIdx.y * 64;

    // staging coords: 8 contiguous elements per thread, 64 rows x 32 cols
    const int srow = tid >> 2;          // 0..63
    const int scol = (tid & 3) * 8;     // 0,8,16,24

    const int nsteps = K >> 5;
    v8f acc0 = {}, acc1 = {};

    // ---- stage one K-tile into buffer `buf` ----
    auto stage = [&](int ks, int buf) {
        int k0 = ks << 5;
        // weight tile: async DMA global->LDS, 16B per lane, contiguous both sides
        unsigned int       loff = (unsigned int)(uintptr_t)&ldsW[buf][srow][scol];
        unsigned long long ga   = (unsigned long long)(uintptr_t)
                                  &Wq[(size_t)(nbase + srow) * K + k0 + scol];
        asm volatile("global_load_async_to_lds_b128 %0, %1, off"
                     :: "v"(loff), "v"(ga) : "memory");
        // A tile: f32 global -> cvt -> f16 LDS (2x float4 -> 1x ds_store_b128)
        const float* ap = &A[(size_t)(rbase + srow) * K + k0 + scol];
        float4 f0 = *(const float4*)ap;
        float4 f1 = *(const float4*)(ap + 4);
        v8h hv;
        hv[0] = (_Float16)f0.x; hv[1] = (_Float16)f0.y;
        hv[2] = (_Float16)f0.z; hv[3] = (_Float16)f0.w;
        hv[4] = (_Float16)f1.x; hv[5] = (_Float16)f1.y;
        hv[6] = (_Float16)f1.z; hv[7] = (_Float16)f1.w;
        *(v8h*)&ldsA[buf][srow][scol] = hv;
    };

    stage(0, 0);

    for (int ks = 0; ks < nsteps; ++ks) {
        int cur = ks & 1;
        asm volatile("s_wait_asynccnt 0x0" ::: "memory");  // DMA for tile ks done
        __syncthreads();                                    // all staging visible
        if (ks + 1 < nsteps) stage(ks + 1, cur ^ 1);        // prefetch next tile

        // ---- fragments from buffer `cur` (all contiguous b128 LDS loads) ----
        v16h af0, af1, bf;
        const int mrow = tm * 16 + l16;
#pragma unroll
        for (int v = 0; v < 8; ++v) {
#pragma unroll
            for (int p = 0; p < 2; ++p) {
                int j  = v * 2 + p;
                int kl = (v < 4) ? (half * 8 + v * 2 + p)
                                 : (16 + half * 8 + (v - 4) * 2 + p);
                af0[j] = ldsA[cur][mrow][kl];
                af1[j] = ldsA[cur][mrow + 32][kl];
            }
        }
        const int ncol = tn * 16 + l16;
#pragma unroll
        for (int j = 0; j < 16; ++j)
            bf[j] = ldsW[cur][ncol][(j < 8) ? (half * 8 + j) : (16 + half * 8 + (j - 8))];

        acc0 = __builtin_amdgcn_wmma_f32_16x16x32_f16(
                   false, af0, false, bf, (short)0, acc0, false, false);
        acc1 = __builtin_amdgcn_wmma_f32_16x16x32_f16(
                   false, af1, false, bf, (short)0, acc1, false, false);
    }

    // ---- epilogue (branch-free per template): VGPR i -> M=i / M=8+i ----
    const int n  = nbase + tn * 16 + l16;
    const int r0 = rbase + tm * 16 + half * 8;
    float add0[8], add1[8];
    if (HAS_ADD) {
#pragma unroll
        for (int i = 0; i < 8; ++i) {
            add0[i] = addend[(size_t)(r0 + i) * Ncols + n];
            add1[i] = addend[(size_t)(r0 + 32 + i) * Ncols + n];
        }
    }
#pragma unroll
    for (int i = 0; i < 8; ++i) {
        float v0 = acc0[i], v1 = acc1[i];
        if (HAS_ADD)  { v0 += add0[i]; v1 += add1[i]; }
        if (DO_QUANT) { v0 = qdev(v0); v1 = qdev(v1); }
        Cout[(size_t)(r0 + i) * Ncols + n]      = v0;
        Cout[(size_t)(r0 + 32 + i) * Ncols + n] = v1;
    }
}

// ---- sequential scan (in place): S[l] = q(Aq * S[l-1] + V[l]), chain per (b,n) ----
__global__ void scan_kernel(float* __restrict__ V, const float* __restrict__ Aq) {
    int c = blockIdx.x * blockDim.x + threadIdx.x;  // 0..16383
    int b = c >> 9;
    int n = c & (NN - 1);
    float a = Aq[n];
    float s = 0.0f;
    for (int l = 0; l < LEN; ++l) {
        size_t idx = ((size_t)l * BZ + b) * NN + n;
        s = qdev(fmaf(a, s, V[idx]));
        V[idx] = s;
    }
}

// ---- Ytmp [(l*BZ+b)][dout] f32 -> out[b][dout][l] ----
__global__ void transpose_y_kernel(const float* __restrict__ Ytmp, float* __restrict__ out) {
    __shared__ float tile[32][33];
    int l0 = blockIdx.x * 32, d0 = blockIdx.y * 32, b = blockIdx.z;
    int tx = threadIdx.x, ty = threadIdx.y;            // (32, 8)
#pragma unroll
    for (int i = 0; i < 4; ++i) {
        int l = l0 + ty + i * 8;
        tile[ty + i * 8][tx] = Ytmp[((size_t)l * BZ + b) * DOUT + d0 + tx];
    }
    __syncthreads();
    float* ob = out + (size_t)b * DOUT * LEN;
#pragma unroll
    for (int i = 0; i < 4; ++i) {
        int d = d0 + ty + i * 8;
        ob[(size_t)d * LEN + l0 + tx] = tile[tx][ty + i * 8];
    }
}

extern "C" void kernel_launch(void* const* d_in, const int* in_sizes, int n_in,
                              void* d_out, int out_size, void* d_ws, size_t ws_size,
                              hipStream_t stream) {
    const float* u  = (const float*)d_in[0];   // (32, 512, 512)
    const float* Ad = (const float*)d_in[1];   // (512,)
    const float* B  = (const float*)d_in[2];   // (512, 512)
    const float* C  = (const float*)d_in[3];   // (512, 512)
    const float* D  = (const float*)d_in[4];   // (512, 512)
    float* out = (float*)d_out;

    char* ws = (char*)d_ws;
    size_t szBig = (size_t)RR * DIN * sizeof(float);          // 33.5 MB
    float*    Urows = (float*)(ws);                            // later reused as Ytmp
    float*    V     = (float*)(ws + szBig);                    // later overwritten with S
    float*    W     = (float*)(ws + 2 * szBig);
    _Float16* Bq    = (_Float16*)(ws + 3 * szBig);             // (NN   x DIN) row-major
    _Float16* Cq    = (_Float16*)(Bq + (size_t)NN * DIN);      // (DOUT x NN)  row-major
    _Float16* Dq    = (_Float16*)(Cq + (size_t)DOUT * NN);     // (DOUT x DIN) row-major
    float*    Aq    = (float*)(Dq + (size_t)DOUT * DIN);
    float*    Ytmp  = Urows;

    // 1) quantize weights elementwise to f16 (layouts already [n][k] for the GEMMs)
    {
        int nEl = NN * DIN;
        quant_kernel<<<(nEl + 255) / 256, 256, 0, stream>>>(B, Bq, nEl);
        quant_kernel<<<(nEl + 255) / 256, 256, 0, stream>>>(C, Cq, nEl);
        quant_kernel<<<(nEl + 255) / 256, 256, 0, stream>>>(D, Dq, nEl);
        quantA_kernel<<<(NN + 255) / 256, 256, 0, stream>>>(Ad, Aq, NN);
    }

    // 2) u -> Urows [(l*BZ+b)][d]
    {
        dim3 grid(LEN / 32, DIN / 32, BZ), block(32, 8);
        transpose_u_kernel<<<grid, block, 0, stream>>>(u, Urows);
    }

    // 3) V = Urows @ B_q^T ; W = Urows @ D_q^T   (WMMA f16 -> f32)
    {
        dim3 grid(RR / 64, NN / 64), block(256);
        wmma_gemm_kernel<false, false><<<grid, block, 0, stream>>>(Urows, Bq, V, nullptr, DIN, NN);
        wmma_gemm_kernel<false, false><<<grid, block, 0, stream>>>(Urows, Dq, W, nullptr, DIN, DOUT);
    }

    // 4) in-place scan: V becomes S
    scan_kernel<<<(BZ * NN) / 256, 256, 0, stream>>>(V, Aq);

    // 5) Ytmp = q(S @ C_q^T + W)   (fused epilogue)
    {
        dim3 grid(RR / 64, DOUT / 64), block(256);
        wmma_gemm_kernel<true, true><<<grid, block, 0, stream>>>(V, Cq, Ytmp, W, NN, DOUT);
    }

    // 6) Ytmp -> out (b, dout, l)
    {
        dim3 grid(LEN / 32, DOUT / 32, BZ), block(32, 8);
        transpose_y_kernel<<<grid, block, 0, stream>>>(Ytmp, out);
    }
}
